// GnnActor_14276471292240
// MI455X (gfx1250) — compile-verified, matrix operand order checked by reading, not run on hardware
//
#include <hip/hip_runtime.h>

#define BATCH 8192
#define DBODY 64
#define DOBJ 128
#define NOBJ 9
#define NEDGE 72
#define DEDGE 128
#define NISO 3
#define NNODE 12
#define DIN 320
#define HID 256
#define ACT 32
#define TB 32

#define S_IN 328   // LDS row stride (ushorts) for 320-wide input tile (+8 pad, conflict-free)
#define S_H  264   // LDS row stride for 256-wide h / pooled / r tiles (+8 pad)

typedef __attribute__((ext_vector_type(16))) __bf16          v16bf;
typedef __attribute__((ext_vector_type(16))) unsigned short  v16u;
typedef __attribute__((ext_vector_type(8)))  unsigned int    v8u;
typedef __attribute__((ext_vector_type(8)))  float           v8f;

// Native f32 -> bf16 (RNE) via hardware convert.
__device__ __forceinline__ unsigned short f2bf(float x) {
  __bf16 h = (__bf16)x;
  return __builtin_bit_cast(unsigned short, h);
}

__device__ __forceinline__ v8f wmma_bf16(v16bf a, v16bf b, v8f c) {
  // D(f32 16x16) = A(bf16 16x32) * B(bf16 32x16) + C
  return __builtin_amdgcn_wmma_f32_16x16x32_bf16(false, a, false, b, (short)0, c, false, false);
}

// A-fragment (16x32 bf16) from row-major LDS tile.
// Lane 0-15: M=lane, K = k0+{0..7,16..23}; lane 16-31: M=lane-16, K = k0+{8..15,24..31}.
__device__ __forceinline__ v16bf load_a_frag(const unsigned short* base, int stride,
                                             int m0, int k0, int lane) {
  const int m  = m0 + (lane & 15);
  const int bk = k0 + ((lane & 16) ? 8 : 0);
  const unsigned short* p = base + m * stride + bk;
  v16u u;
#pragma unroll
  for (int j = 0; j < 4; ++j) {
    const unsigned d = *(const unsigned*)(p + 2 * j);
    u[2 * j]     = (unsigned short)d;
    u[2 * j + 1] = (unsigned short)(d >> 16);
  }
#pragma unroll
  for (int j = 0; j < 4; ++j) {
    const unsigned d = *(const unsigned*)(p + 16 + 2 * j);
    u[8 + 2 * j] = (unsigned short)d;
    u[9 + 2 * j] = (unsigned short)(d >> 16);
  }
  return __builtin_bit_cast(v16bf, u);
}

// B-fragment (32x16 bf16) from pre-swizzled packed weights: 512 contiguous bf16 per fragment,
// lane's 16 values contiguous -> two global_load_b128.
__device__ __forceinline__ v16bf load_b_frag(const unsigned short* wp, int fragIdx, int lane) {
  const v8u* p = (const v8u*)(wp + fragIdx * 512 + lane * 16);
  v8u v = *p;
  return __builtin_bit_cast(v16bf, v);
}

// Store a wave's 32(M) x 32(N) f32 accumulator tile to LDS as bf16 with bias+ReLU.
// C/D layout: lane l, vgpr r -> (m = (l>>4)*8 + r, n = l&15) within each 16x16 fragment.
__device__ __forceinline__ void store_relu_bf16(unsigned short* dst, int stride,
                                                const v8f (&acc)[2][2], const float* bias,
                                                int wave, int lane) {
  const int nlo = lane & 15;
  const int hi8 = (lane >> 4) * 8;
#pragma unroll
  for (int mf = 0; mf < 2; ++mf)
#pragma unroll
    for (int f = 0; f < 2; ++f) {
      const int n = wave * 32 + f * 16 + nlo;
#pragma unroll
      for (int r = 0; r < 8; ++r) {
        const int m = mf * 16 + hi8 + r;
        dst[m * stride + n] = f2bf(fmaxf(acc[mf][f][r] + bias[f], 0.0f));
      }
    }
}

// ---------------------------------------------------------------------------
// Kernel 0: convert f32 weights to bf16 pre-swizzled into WMMA B-fragment order.
// Packed element t = fragIdx*512 + lane*16 + j ; fragIdx = (k0/32)*(N/16) + (n0/16);
// lane holds column n0+(lane&15); j<8 -> K=bk+j, j>=8 -> K=bk+16+(j-8), bk=(lane&16)?8:0.
// src2 != nullptr => head mode: cols [0,32) from src (Wm), [32,64) from src2 (Ws), both K x 32.
// ---------------------------------------------------------------------------
__global__ void pack_weights(const float* __restrict__ src, const float* __restrict__ src2,
                             int K, int N, unsigned short* __restrict__ dst) {
  const int t = blockIdx.x * 256 + threadIdx.x;
  if (t >= K * N) return;
  const int Nf   = N >> 4;
  const int fi   = t >> 9;
  const int r    = t & 511;
  const int lane = r >> 4;
  const int j    = r & 15;
  const int k0   = (fi / Nf) << 5;
  const int n0   = (fi % Nf) << 4;
  const int n    = n0 + (lane & 15);
  const int kk   = k0 + ((lane & 16) ? 8 : 0) + ((j < 8) ? j : (j + 8));
  float v;
  if (src2 != nullptr) {
    v = (n < 32) ? src[kk * 32 + n] : src2[kk * 32 + (n - 32)];
  } else {
    v = src[kk * N + n];
  }
  dst[t] = f2bf(v);
}

// ---------------------------------------------------------------------------
// Kernel 1: fully fused GNN actor. One block = 32 batch rows, 8 waves.
// ---------------------------------------------------------------------------
__global__ __launch_bounds__(256)
void gnn_actor_main(const float* __restrict__ obs,
                    const float* __restrict__ ef,
                    const int*   __restrict__ edges_to,
                    const float* __restrict__ iso,
                    const float* __restrict__ isof,
                    const float* __restrict__ b1,
                    const float* __restrict__ b2,
                    const float* __restrict__ br,
                    const float* __restrict__ bm,
                    const float* __restrict__ bs,
                    const unsigned short* __restrict__ W1p,
                    const unsigned short* __restrict__ W2p,
                    const unsigned short* __restrict__ Wrp,
                    const unsigned short* __restrict__ Whp,
                    float* __restrict__ out) {
  __shared__ unsigned short lds_in[TB * S_IN];   // input tile (bf16), 32 x 320
  __shared__ unsigned short lds_h [TB * S_H];    // h / r tile (bf16), 32 x 256
  __shared__ unsigned short lds_p [TB * S_H];    // pooled tile (bf16), 32 x 256
  __shared__ int elist[NOBJ * NEDGE];
  __shared__ int ecnt[NOBJ];

  const int tid  = threadIdx.x;
  const int wave = tid >> 5;
  const int lane = tid & 31;
  const int nlo  = lane & 15;
  const int b0   = blockIdx.x * TB;

  // Staging thread mapping (also used before the node loop for the body part).
  const int srow = tid >> 3;      // 0..31 batch row within tile
  const int sg   = tid & 7;       // column group
  const long sb  = b0 + srow;
  const float* obsrow = obs + sb * (long)(DBODY + NOBJ * DOBJ);

  // Build per-object edge lists from edges_to (general segment_max support).
  if (tid == 0) {
#pragma unroll
    for (int o = 0; o < NOBJ; ++o) ecnt[o] = 0;
    for (int e = 0; e < NEDGE; ++e) {
      const int o = edges_to[e];
      if (o >= 0 && o < NOBJ) { elist[o * NEDGE + ecnt[o]] = e; ecnt[o]++; }
    }
  }

  // Body columns (0..63) are node-invariant: stage them once.
  {
    unsigned short* dst = lds_in + srow * S_IN;
    const float4* p4 = (const float4*)(obsrow + sg * 8);
    float4 v0 = p4[0], v1 = p4[1];
    dst[sg * 8 + 0] = f2bf(v0.x); dst[sg * 8 + 1] = f2bf(v0.y);
    dst[sg * 8 + 2] = f2bf(v0.z); dst[sg * 8 + 3] = f2bf(v0.w);
    dst[sg * 8 + 4] = f2bf(v1.x); dst[sg * 8 + 5] = f2bf(v1.y);
    dst[sg * 8 + 6] = f2bf(v1.z); dst[sg * 8 + 7] = f2bf(v1.w);
  }

  // Per-lane biases for this wave's 32 output columns (n = wave*32 + f*16 + nlo).
  float bias1[2], bias2[2], biasr[2];
#pragma unroll
  for (int f = 0; f < 2; ++f) {
    const int n = wave * 32 + f * 16 + nlo;
    bias1[f] = b1[n];
    bias2[f] = b2[n];
    biasr[f] = br[n];
  }

  const v8f vzero = {0.f, 0.f, 0.f, 0.f, 0.f, 0.f, 0.f, 0.f};
  v8f pooled[2][2];
#pragma unroll
  for (int mf = 0; mf < 2; ++mf)
#pragma unroll
    for (int f = 0; f < 2; ++f) pooled[mf][f] = vzero;

  __syncthreads();   // elist + body tile visible

  for (int node = 0; node < NNODE; ++node) {
    // ---- Stage varying input columns: [node_vec(128) | features(128)] as bf16 ----
    {
      unsigned short* dst = lds_in + srow * S_IN;

      // node vector: 16 floats per thread at cols 64 + sg*16
      {
        const float* np = (node < NOBJ)
                            ? (obsrow + DBODY + node * DOBJ)
                            : (iso + (sb * NISO + (node - NOBJ)) * (long)DOBJ);
        const float4* q4 = (const float4*)(np + sg * 16);
#pragma unroll
        for (int q = 0; q < 4; ++q) {
          float4 t = q4[q];
          unsigned short* d = dst + 64 + sg * 16 + 4 * q;
          d[0] = f2bf(t.x); d[1] = f2bf(t.y); d[2] = f2bf(t.z); d[3] = f2bf(t.w);
        }
      }
      // features: segment-max over incoming edges (connected) or iso features
      if (node < NOBJ) {
        float mx[16];
#pragma unroll
        for (int j = 0; j < 16; ++j) mx[j] = -__builtin_inff();
        const int cnt  = ecnt[node];
        const int next = node + 1;
        const int ncnt = (next < NOBJ) ? ecnt[next] : 0;
        for (int i = 0; i < cnt; ++i) {
          const int e = elist[node * NEDGE + i];
          const float4* e4 = (const float4*)(ef + ((sb * NEDGE) + e) * (long)DEDGE + sg * 16);
          // Prefetch the matching edge row of the next node while this one streams in.
          if (i < ncnt) {
            const int en = elist[next * NEDGE + i];
            __builtin_prefetch(ef + ((sb * NEDGE) + en) * (long)DEDGE + sg * 16, 0, 0);
          }
#pragma unroll
          for (int q = 0; q < 4; ++q) {
            float4 t = e4[q];
            mx[4 * q + 0] = fmaxf(mx[4 * q + 0], t.x);
            mx[4 * q + 1] = fmaxf(mx[4 * q + 1], t.y);
            mx[4 * q + 2] = fmaxf(mx[4 * q + 2], t.z);
            mx[4 * q + 3] = fmaxf(mx[4 * q + 3], t.w);
          }
        }
#pragma unroll
        for (int j = 0; j < 16; ++j) dst[192 + sg * 16 + j] = f2bf(mx[j]);
      } else {
        const float4* f4 = (const float4*)(isof + (sb * NISO + (node - NOBJ)) * (long)DEDGE + sg * 16);
#pragma unroll
        for (int q = 0; q < 4; ++q) {
          float4 t = f4[q];
          unsigned short* d = dst + 192 + sg * 16 + 4 * q;
          d[0] = f2bf(t.x); d[1] = f2bf(t.y); d[2] = f2bf(t.z); d[3] = f2bf(t.w);
        }
      }
    }
    __syncthreads();

    // ---- Layer 1: h = relu(inp(32x320) @ W1(320x256) + b1) ----
    v8f acc[2][2];
#pragma unroll
    for (int mf = 0; mf < 2; ++mf)
#pragma unroll
      for (int f = 0; f < 2; ++f) acc[mf][f] = vzero;

#pragma unroll
    for (int k0 = 0; k0 < DIN; k0 += 32) {
      const v16bf a0 = load_a_frag(lds_in, S_IN, 0, k0, lane);
      const v16bf a1 = load_a_frag(lds_in, S_IN, 16, k0, lane);
#pragma unroll
      for (int f = 0; f < 2; ++f) {
        const v16bf bb = load_b_frag(W1p, (k0 >> 5) * (HID / 16) + wave * 2 + f, lane);
        acc[0][f] = wmma_bf16(a0, bb, acc[0][f]);
        acc[1][f] = wmma_bf16(a1, bb, acc[1][f]);
      }
    }
    store_relu_bf16(lds_h, S_H, acc, bias1, wave, lane);
    __syncthreads();

    // ---- Layer 2: phi = relu(h(32x256) @ W2(256x256) + b2); pooled += phi ----
#pragma unroll
    for (int mf = 0; mf < 2; ++mf)
#pragma unroll
      for (int f = 0; f < 2; ++f) acc[mf][f] = vzero;

#pragma unroll
    for (int k0 = 0; k0 < HID; k0 += 32) {
      const v16bf a0 = load_a_frag(lds_h, S_H, 0, k0, lane);
      const v16bf a1 = load_a_frag(lds_h, S_H, 16, k0, lane);
#pragma unroll
      for (int f = 0; f < 2; ++f) {
        const v16bf bb = load_b_frag(W2p, (k0 >> 5) * (HID / 16) + wave * 2 + f, lane);
        acc[0][f] = wmma_bf16(a0, bb, acc[0][f]);
        acc[1][f] = wmma_bf16(a1, bb, acc[1][f]);
      }
    }
#pragma unroll
    for (int mf = 0; mf < 2; ++mf)
#pragma unroll
      for (int f = 0; f < 2; ++f)
#pragma unroll
        for (int r = 0; r < 8; ++r)
          pooled[mf][f][r] += fmaxf(acc[mf][f][r] + bias2[f], 0.0f);
    __syncthreads();
  }

  // ---- Write pooled (raw, no relu) to LDS as bf16 for the rho layer ----
  {
    const int hi8 = (lane >> 4) * 8;
#pragma unroll
    for (int mf = 0; mf < 2; ++mf)
#pragma unroll
      for (int f = 0; f < 2; ++f) {
        const int n = wave * 32 + f * 16 + nlo;
#pragma unroll
        for (int r = 0; r < 8; ++r)
          lds_p[(mf * 16 + hi8 + r) * S_H + n] = f2bf(pooled[mf][f][r]);
      }
  }
  __syncthreads();

  // ---- Rho: r = relu(pooled(32x256) @ W1_rho(256x256) + br) ----
  {
    v8f acc[2][2];
#pragma unroll
    for (int mf = 0; mf < 2; ++mf)
#pragma unroll
      for (int f = 0; f < 2; ++f) acc[mf][f] = vzero;

#pragma unroll
    for (int k0 = 0; k0 < HID; k0 += 32) {
      const v16bf a0 = load_a_frag(lds_p, S_H, 0, k0, lane);
      const v16bf a1 = load_a_frag(lds_p, S_H, 16, k0, lane);
#pragma unroll
      for (int f = 0; f < 2; ++f) {
        const v16bf bb = load_b_frag(Wrp, (k0 >> 5) * (HID / 16) + wave * 2 + f, lane);
        acc[0][f] = wmma_bf16(a0, bb, acc[0][f]);
        acc[1][f] = wmma_bf16(a1, bb, acc[1][f]);
      }
    }
    store_relu_bf16(lds_h, S_H, acc, biasr, wave, lane);
  }
  __syncthreads();

  // ---- Heads: [mean | log_std] = r(32x256) @ [Wm|Ws](256x64); waves 0-3 only ----
  if (wave < 4) {
    v8f ha[2] = {vzero, vzero};
#pragma unroll
    for (int k0 = 0; k0 < HID; k0 += 32) {
      const v16bf a0 = load_a_frag(lds_h, S_H, 0, k0, lane);
      const v16bf a1 = load_a_frag(lds_h, S_H, 16, k0, lane);
      const v16bf bb = load_b_frag(Whp, (k0 >> 5) * 4 + wave, lane);
      ha[0] = wmma_bf16(a0, bb, ha[0]);
      ha[1] = wmma_bf16(a1, bb, ha[1]);
    }
    const int ng   = wave * 16 + nlo;                   // 0..63 combined column
    const float hb = (ng < ACT) ? bm[ng] : bs[ng - ACT];
    const int hi8  = (lane >> 4) * 8;
#pragma unroll
    for (int mf = 0; mf < 2; ++mf)
#pragma unroll
      for (int r = 0; r < 8; ++r) {
        const long b = b0 + mf * 16 + hi8 + r;
        const float v = ha[mf][r] + hb;
        if (ng < ACT) {
          out[b * ACT + ng] = v;                                          // mean
        } else {
          out[(long)BATCH * ACT + b * ACT + (ng - ACT)] =                 // log_std
              fminf(fmaxf(v, -20.0f), 2.0f);
        }
      }
  }
}

extern "C" void kernel_launch(void* const* d_in, const int* in_sizes, int n_in,
                              void* d_out, int out_size, void* d_ws, size_t ws_size,
                              hipStream_t stream) {
  (void)in_sizes; (void)n_in; (void)out_size; (void)ws_size;
  const float* obs      = (const float*)d_in[0];
  const float* ef       = (const float*)d_in[1];
  const int*   edges_to = (const int*)d_in[2];
  const float* iso      = (const float*)d_in[3];
  const float* isof     = (const float*)d_in[4];
  const float* W1       = (const float*)d_in[5];
  const float* b1       = (const float*)d_in[6];
  const float* W2       = (const float*)d_in[7];
  const float* b2       = (const float*)d_in[8];
  const float* Wr       = (const float*)d_in[9];
  const float* br       = (const float*)d_in[10];
  const float* Wm       = (const float*)d_in[11];
  const float* bm       = (const float*)d_in[12];
  const float* Ws       = (const float*)d_in[13];
  const float* bs       = (const float*)d_in[14];
  float* out = (float*)d_out;

  unsigned short* wsp = (unsigned short*)d_ws;
  unsigned short* W1p = wsp;                      // 320*256
  unsigned short* W2p = W1p + 320 * 256;          // 256*256
  unsigned short* Wrp = W2p + 256 * 256;          // 256*256
  unsigned short* Whp = Wrp + 256 * 256;          // 256*64

  pack_weights<<<(320 * 256 + 255) / 256, 256, 0, stream>>>(W1, nullptr, 320, 256, W1p);
  pack_weights<<<(256 * 256 + 255) / 256, 256, 0, stream>>>(W2, nullptr, 256, 256, W2p);
  pack_weights<<<(256 * 256 + 255) / 256, 256, 0, stream>>>(Wr, nullptr, 256, 256, Wrp);
  pack_weights<<<(256 * 64 + 255) / 256, 256, 0, stream>>>(Wm, Ws, 256, 64, Whp);

  gnn_actor_main<<<BATCH / TB, 256, 0, stream>>>(obs, ef, edges_to, iso, isof,
                                                 b1, b2, br, bm, bs,
                                                 W1p, W2p, Wrp, Whp, out);
}
